// BetaVAEMark10Decoder_60224031425238
// MI455X (gfx1250) — matrix operand order repaired and checked by reading, not compiled, not run on hardware
//
#include <hip/hip_runtime.h>

// CDNA5 / gfx1250 fused BetaVAE decoder.
// Kernel 1 (prep): convert/swizzle all weights to bf16 once into d_ws:
//   - 3x3 conv weights pre-laid-out as per-lane WMMA B fragments
//   - conv-transpose weights transposed to [tap][co][ci] for contiguous reads
// Kernel 2 (main): one workgroup (256 thr = 8 wave32) per image; activations in
// LDS as bf16 with zero halos; 3x3 SAME convs are implicit GEMM on
// v_wmma_f32_16x16x32_bf16 with b128 LDS gathers and b128 B-fragment loads.

typedef __attribute__((ext_vector_type(16))) __bf16       v16bf;
typedef __attribute__((ext_vector_type(8)))  float        v8f;
typedef __attribute__((ext_vector_type(4)))  unsigned int v4u;
typedef __attribute__((ext_vector_type(4)))  float        v4f;

#define NEG_SLOPE 0.01f

__device__ __forceinline__ unsigned short f2bf(float f) {
  unsigned int u = __float_as_uint(f);
  unsigned int r = u + 0x7FFFu + ((u >> 16) & 1u);   // round-to-nearest-even
  return (unsigned short)(r >> 16);
}
// bf16 pair packed in a dword -> floats (bf16 == high half of f32)
__device__ __forceinline__ float bfu_lo(unsigned int u) { return __uint_as_float(u << 16); }
__device__ __forceinline__ float bfu_hi(unsigned int u) { return __uint_as_float(u & 0xFFFF0000u); }

union Frag32B { v16bf v; v4u q[2]; };                // 32B: two b128 chunks

// ---------------- d_ws layout (ushort offsets) -----------------------------
// B-fragments:  B1 @0     (9*32*16 = 4608)   B2 @4608 (5*32*16 = 2560)
//               B3 @7168  (3*32*16 = 1536)
// up weights^T: U1 @8704  (10*32*32 = 10240) U2 @18944 (10*16*16 = 2560)
//               U3 @21504 (4*8*8 = 256)      total 21760 ushorts = 43520 B
#define WS_B1 0
#define WS_B2 4608
#define WS_B3 7168
#define WS_U1 8704
#define WS_U2 18944
#define WS_U3 21504

// ---------------------------------------------------------------------------
// Prep: swizzle 3x3 conv weights (3,3,Cin,Cout) HWIO f32 into per-lane WMMA
// B-fragment order: dst[(s*32 + lane)*16 + e] holds K = s*32 + (lane>=16)*16+e,
// N = lane&15; zero for K >= 9*Cin or N >= Cout.
// ---------------------------------------------------------------------------
template<int Cin, int Cout, int STEPS>
__device__ void prep_bfrag(const float* __restrict__ Wg,
                           unsigned short* __restrict__ dst, int tid, int nt) {
  constexpr int L2C = (Cin == 32) ? 5 : (Cin == 16) ? 4 : 3;
  for (int i = tid; i < STEPS * 32 * 16; i += nt) {
    int e = i & 15, lane = (i >> 4) & 31, s = i >> 9;
    int n = lane & 15, khalf = lane >> 4;
    int k = s * 32 + khalf * 16 + e;
    int tap = k >> L2C, ci = k & (Cin - 1);
    unsigned short v = 0;
    if (tap < 9 && n < Cout) v = f2bf(Wg[(tap * Cin + ci) * Cout + n]);
    dst[i] = v;
  }
}
// Transpose (KH*KW, Cin=C, Cout=C) HWIO f32 -> [tap][co][ci] bf16.
template<int C, int TAPS>
__device__ void prep_upT(const float* __restrict__ Wg,
                         unsigned short* __restrict__ dst, int tid, int nt) {
  for (int i = tid; i < TAPS * C * C; i += nt) {
    int ci = i % C, co = (i / C) % C, t = i / (C * C);
    dst[i] = f2bf(Wg[(t * C + ci) * C + co]);
  }
}

__global__ __launch_bounds__(256)
void prep_weights(const float* __restrict__ Wup1, const float* __restrict__ Wc1,
                  const float* __restrict__ Wup2, const float* __restrict__ Wc2,
                  const float* __restrict__ Wup3, const float* __restrict__ Wc3,
                  unsigned short* __restrict__ ws) {
  const int tid = blockIdx.x * 256 + threadIdx.x;
  const int nt  = gridDim.x * 256;
  prep_bfrag<32, 16, 9>(Wc1, ws + WS_B1, tid, nt);
  prep_bfrag<16,  8, 5>(Wc2, ws + WS_B2, tid, nt);
  prep_bfrag< 8,  6, 3>(Wc3, ws + WS_B3, tid, nt);
  prep_upT<32, 10>(Wup1, ws + WS_U1, tid, nt);
  prep_upT<16, 10>(Wup2, ws + WS_U2, tid, nt);
  prep_upT< 8,  4>(Wup3, ws + WS_U3, tid, nt);
}

// ---------------------------------------------------------------------------
// 3x3 SAME conv as implicit GEMM on WMMA bf16.
//   actIn : LDS bf16, halo NHWC (H+2)x(W+2)xCin, border pre-zeroed.
//   Bpre  : pre-swizzled bf16 B fragments in global (L2-resident).
// ---------------------------------------------------------------------------
template<int H, int W, int Cin, int Cout, bool FINAL>
__device__ void conv3x3_same_wmma(const unsigned short* __restrict__ actIn,
                                  unsigned short* __restrict__ actOut,   // plain NHWC
                                  const unsigned short* __restrict__ Bpre,
                                  float* __restrict__ outG, int img) {
  constexpr int L2C   = (Cin == 32) ? 5 : (Cin == 16) ? 4 : 3;
  constexpr int STEPS = (9 * Cin + 31) / 32;
  constexpr int MT    = (H * W) / 16;
  constexpr int WH    = W + 2;                       // halo row pitch (pixels)
  const int lane  = threadIdx.x & 31;
  const int wave  = threadIdx.x >> 5;
  const int n     = lane & 15;
  const int khalf = lane >> 4;

  // B fragments: 2 x global_load_b128 per step, no conversion.
  Frag32B bfrag[STEPS];
#pragma unroll
  for (int s = 0; s < STEPS; ++s) {
    const v4u* p = (const v4u*)(Bpre + (s * 32 + lane) * 16);
    bfrag[s].q[0] = p[0];
    bfrag[s].q[1] = p[1];
  }

  for (int t = wave; t < MT; t += 8) {               // wave-uniform: EXEC all-ones at WMMA
    const int pos = t * 16 + n;                      // A row: M = lane & 15
    const int h0 = pos / W, w0 = pos % W;
    v8f acc = {};
#pragma unroll
    for (int s = 0; s < STEPS; ++s) {
      Frag32B A;
      // Chunk q = 8 consecutive K: one tap, contiguous ci -> aligned ds_load_b128.
      // Tap clamped for K >= 9*Cin (B is zero there).
#pragma unroll
      for (int q = 0; q < 2; ++q) {
        int k0  = s * 32 + khalf * 16 + (q << 3);
        int tap = k0 >> L2C;
        tap     = tap > 8 ? 8 : tap;
        int ci  = k0 & (Cin - 1);
        int t3  = tap / 3;
        int row = h0 + t3;                           // +1 halo folded in
        int col = w0 + (tap - 3 * t3);
        A.q[q]  = *(const v4u*)(actIn + (row * WH + col) * Cin + ci);
      }
      acc = __builtin_amdgcn_wmma_f32_16x16x32_bf16(
          false, A.v, false, bfrag[s].v, (short)0, acc, false, false);
    }
    // D layout: VGPR r -> M = r + 8*khalf, N = lane&15
    if (FINAL) {
      if (n < Cout) {
        int opos0 = t * 16 + 8 * khalf;              // 8 consecutive positions
        int oh = opos0 / W, ow = opos0 % W;          // tiles never cross rows (16|W)
        float* dst = outG + (((long)img * Cout + n) * H + oh) * W + ow;
        v4f a0, a1;
#pragma unroll
        for (int r = 0; r < 4; ++r) {
          float v0 = acc[r];     a0[r] = v0 > 0.0f ? v0 : 0.0f;
          float v1 = acc[r + 4]; a1[r] = v1 > 0.0f ? v1 : 0.0f;
        }
        *(v4f*)(dst)     = a0;                       // 32B-aligned NCHW runs
        *(v4f*)(dst + 4) = a1;
      }
    } else {
      if (n < Cout) {
#pragma unroll
        for (int r = 0; r < 8; ++r) {
          int opos = t * 16 + 8 * khalf + r;
          float v  = acc[r];
          v = v > 0.0f ? v : NEG_SLOPE * v;
          actOut[opos * Cout + n] = f2bf(v);
        }
      }
    }
  }
}

// ---------------------------------------------------------------------------
// Conv-transpose, stride == kernel: out(oh,ow) <- in(oh/KH, ow/KW) through tap
// (KH-1-oh%KH, KW-1-ow%KW)  (JAX conv_transpose, transpose_kernel=False).
// Weights pre-transposed bf16 [tap][co][ci]; b128 loads on both operands.
// ---------------------------------------------------------------------------
template<int IH, int IW, int C, int KH, int KW>
__device__ void up_dense_halo(const unsigned short* __restrict__ in,
                              unsigned short* __restrict__ outH,   // (OH+2)(OW+2)C halo
                              const unsigned short* __restrict__ Wt) {
  constexpr int OH = IH * KH, OW = IW * KW;
  for (int idx = threadIdx.x; idx < OH * OW * C; idx += 256) {
    int co = idx % C;
    int ow = (idx / C) % OW;
    int oh = idx / (C * OW);
    int ih = oh / KH, iw = ow / KW;
    int kh = (KH - 1) - (oh % KH);
    int kw = (KW - 1) - (ow % KW);
    const v4u* ip = (const v4u*)(in + (ih * IW + iw) * C);          // 16B-aligned
    const v4u* wp = (const v4u*)(Wt + ((kh * KW + kw) * C + co) * C);
    float a = 0.0f;
#pragma unroll
    for (int b = 0; b < C / 8; ++b) {
      v4u d = ip[b];
      v4u w = wp[b];
#pragma unroll
      for (int j = 0; j < 4; ++j) {
        a += bfu_lo(d[j]) * bfu_lo(w[j]);
        a += bfu_hi(d[j]) * bfu_hi(w[j]);
      }
    }
    outH[((oh + 1) * (OW + 2) + (ow + 1)) * C + co] = f2bf(a);
  }
}

template<int NELEM>
__device__ void zero_lds(unsigned short* p) {
  unsigned int* p32 = (unsigned int*)p;
  for (int i = threadIdx.x; i < NELEM / 2; i += 256) p32[i] = 0u;
}

// ---------------------------------------------------------------------------
// Fused decoder, one image per block. Dynamic LDS arena, 67712 bytes:
//  x0  @0      (256 el)        x1h @u29824 (7*18*32  = 4032 el)
//  y1  @0      (1280 el)       x2h @u19168 (27*34*16 = 14688 el)
//  y2  @0      (6400 el)       x3h @u6400  (52*66*8  = 27456 el)
// Peak concurrent: y2 + x3h = 67712 B  (<< 320 KB CDNA5 WGP LDS).
// ---------------------------------------------------------------------------
__global__ __launch_bounds__(256)
void decoder_fused(const float* __restrict__ latent, const float* __restrict__ Wlin,
                   const float* __restrict__ blin,
                   const unsigned short* __restrict__ ws,
                   float* __restrict__ out) {
  extern __shared__ unsigned short smem[];
  const int img = blockIdx.x;

  unsigned short* x0  = smem;            // (1,8,32)
  unsigned short* x1h = smem + 29824;    // (5,16,32) + halo
  unsigned short* y1  = smem;            // (5,16,16)
  unsigned short* x2h = smem + 19168;    // (25,32,16) + halo
  unsigned short* y2  = smem;            // (25,32,8)
  unsigned short* x3h = smem + 6400;     // (50,64,8) + halo

  // Stage 0: linear(4->256) + leaky; h.reshape(32,1,8) -> NHWC (1,8,32)
  {
    const int c = threadIdx.x;
    const float* z = latent + (long)img * 4;
    float a = blin[c];
#pragma unroll
    for (int k = 0; k < 4; ++k) a += z[k] * Wlin[k * 256 + c];
    a = a > 0.0f ? a : NEG_SLOPE * a;
    x0[(c & 7) * 32 + (c >> 3)] = f2bf(a);
  }
  __syncthreads();

  zero_lds<4032>(x1h);            __syncthreads();
  up_dense_halo<1, 8, 32, 5, 2>(x0, x1h, ws + WS_U1);   __syncthreads();
  conv3x3_same_wmma<5, 16, 32, 16, false>(x1h, y1, ws + WS_B1, nullptr, img);
  __syncthreads();

  zero_lds<14688>(x2h);           __syncthreads();
  up_dense_halo<5, 16, 16, 5, 2>(y1, x2h, ws + WS_U2);  __syncthreads();
  conv3x3_same_wmma<25, 32, 16, 8, false>(x2h, y2, ws + WS_B2, nullptr, img);
  __syncthreads();

  zero_lds<27456>(x3h);           __syncthreads();
  up_dense_halo<25, 32, 8, 2, 2>(y2, x3h, ws + WS_U3);  __syncthreads();
  conv3x3_same_wmma<50, 64, 8, 6, true>(x3h, nullptr, ws + WS_B3, out, img);
}

extern "C" void kernel_launch(void* const* d_in, const int* in_sizes, int n_in,
                              void* d_out, int out_size, void* d_ws, size_t ws_size,
                              hipStream_t stream) {
  (void)n_in; (void)out_size; (void)ws_size;
  const int B = in_sizes[0] / 4;        // latent_vector is (B,4)
  unsigned short* ws = (unsigned short*)d_ws;   // needs 43520 B of scratch

  prep_weights<<<16, 256, 0, stream>>>(
      (const float*)d_in[3], (const float*)d_in[4], (const float*)d_in[5],
      (const float*)d_in[6], (const float*)d_in[7], (const float*)d_in[8], ws);

  decoder_fused<<<B, 256, 67712, stream>>>(
      (const float*)d_in[0], (const float*)d_in[1], (const float*)d_in[2],
      ws, (float*)d_out);
}
